// Transformer_14688788152484
// MI455X (gfx1250) — matrix operand order repaired and checked by reading, not compiled, not run on hardware
//
#include <hip/hip_runtime.h>
#include <stdint.h>

// ---------------------------------------------------------------------------
// Types for CDNA5 WMMA (wave32): D(f32 16x16) = A(f16 16x32) * B(f16 32x16) + C
// ---------------------------------------------------------------------------
typedef __attribute__((ext_vector_type(16))) _Float16 v16h;
typedef __attribute__((ext_vector_type(8)))  float    v8f;

#define TPB 256

enum { GF_BIAS = 1, GF_RELU = 2, GF_RESID = 4, GF_OUTH = 8, GF_OUTT = 32 };

union Frag128 { uint4 q[2]; v16h v; };   // two 16B LDS chunks -> one fragment

// ---------------------------------------------------------------------------
// Async global -> LDS copy (CDNA5): one instruction moves 16B per lane,
// tracked by ASYNCcnt.  lds_addr = LDS byte offset (low 32 bits of generic
// shared pointer: flat LDS aperture keeps offset in addr[31:0]).
// ---------------------------------------------------------------------------
__device__ __forceinline__ void async_copy_b128(uint32_t lds_addr, const void* gaddr)
{
    asm volatile("global_load_async_to_lds_b128 %0, %1, off"
                 :: "v"(lds_addr), "v"(gaddr) : "memory");
}
__device__ __forceinline__ void wait_async0()
{
    asm volatile("s_wait_asynccnt 0" ::: "memory");
}

// ---------------------------------------------------------------------------
// WMMA GEMM with async-LDS double-buffered staging.
//   C[M,N] = A[M,K] * Bt^T + epilogue,  A f16 [M,K] lda,  Bt f16 [N,K] ldb.
//   Block: 256 threads = 8 waves, computes 128x64 C tile (wave = 32x32, 2x2 WMMA).
//   Requires M%128==0, N%64==0, K%32==0 (true for every GEMM in this model).
//   Epilogue: +bias[n], +resid[m*ldr+n], relu; out f32 (Cf) / f16 (Ch) /
//   f16-transposed (GF_OUTT: Ch[n*ldc+m]).
// ---------------------------------------------------------------------------
__global__ __launch_bounds__(TPB)
void gemm_wmma(const _Float16* __restrict__ A, int lda,
               const _Float16* __restrict__ B, int ldb,
               float* __restrict__ Cf, _Float16* __restrict__ Ch, int ldc,
               const float* __restrict__ bias,
               const float* __restrict__ resid, int ldr,
               int K, int flags)
{
    __shared__ __align__(16) _Float16 sA[2][128 * 32];   // 16 KB
    __shared__ __align__(16) _Float16 sB[2][64 * 32];    //  8 KB

    const int tid  = threadIdx.x;
    const int lane = tid & 31;
    const int wave = tid >> 5;
    const int rowBase = blockIdx.y * 128;
    const int colBase = blockIdx.x * 64;
    const int wm = (wave >> 1) * 32;          // wave row offset in block tile
    const int wn = (wave & 1) * 32;           // wave col offset in block tile
    const int l16 = lane & 15;
    const int g   = lane >> 4;

    const uint32_t sA_base = (uint32_t)(uintptr_t)&sA[0][0];
    const uint32_t sB_base = (uint32_t)(uintptr_t)&sB[0][0];

    // Stage one 32-wide K slab into LDS buffer `buf` via async copies.
    // A slab: 128 rows x 64B  (512 x 16B chunks, 2 per thread)
    // B slab:  64 rows x 64B  (256 x 16B chunks, 1 per thread)
    auto stage = [&](int buf, int k0) {
        const char* gA = (const char*)(A + (size_t)rowBase * lda + k0);
        #pragma unroll
        for (int i = 0; i < 2; ++i) {
            const int idx = tid + i * TPB;
            const int r = idx >> 2, ch = idx & 3;
            async_copy_b128(sA_base + (uint32_t)(buf * 8192 + r * 64 + ch * 16),
                            gA + (size_t)r * lda * 2 + ch * 16);
        }
        const char* gB = (const char*)(B + (size_t)colBase * ldb + k0);
        {
            const int r = tid >> 2, ch = tid & 3;
            async_copy_b128(sB_base + (uint32_t)(buf * 4096 + r * 64 + ch * 16),
                            gB + (size_t)r * ldb * 2 + ch * 16);
        }
    };

    v8f c[2][2];
    const v8f zero = {};
    c[0][0] = zero; c[0][1] = zero; c[1][0] = zero; c[1][1] = zero;

    stage(0, 0);
    const int ksteps = K >> 5;
    for (int kt = 0; kt < ksteps; ++kt) {
        wait_async0();          // this wave's copies into buf (kt&1) landed
        __syncthreads();        // every wave's copies landed; prev reads done
        if (kt + 1 < ksteps) stage((kt + 1) & 1, (kt + 1) << 5);

        const int buf = kt & 1;
        // A fragments: lane row m = wm + i*16 + l16; per ISA layout the lane's
        // 16 K-slots are two contiguous 16B chunks at +g*16 and +32+g*16.
        v16h a[2], b[2];
        #pragma unroll
        for (int i = 0; i < 2; ++i) {
            const char* rp = (const char*)&sA[buf][(wm + i * 16 + l16) * 32];
            Frag128 f;
            f.q[0] = *(const uint4*)(rp + g * 16);
            f.q[1] = *(const uint4*)(rp + 32 + g * 16);
            a[i] = f.v;
        }
        // B fragments: lane col n = wn + j*16 + l16; same K-slots as A: two
        // 16B chunks at +g*16 and +32+g*16.
        #pragma unroll
        for (int j = 0; j < 2; ++j) {
            const char* rp = (const char*)&sB[buf][(wn + j * 16 + l16) * 32];
            Frag128 f;
            f.q[0] = *(const uint4*)(rp + g * 16);
            f.q[1] = *(const uint4*)(rp + 32 + g * 16);
            b[j] = f.v;
        }
        #pragma unroll
        for (int i = 0; i < 2; ++i)
            #pragma unroll
            for (int j = 0; j < 2; ++j)
                c[i][j] = __builtin_amdgcn_wmma_f32_16x16x32_f16(
                    false, a[i], false, b[j], (short)0, c[i][j], false, false);
    }

    // Epilogue: C layout: reg r, lane-group g -> m = r + 8g, n = l16.
    #pragma unroll
    for (int i = 0; i < 2; ++i) {
        #pragma unroll
        for (int j = 0; j < 2; ++j) {
            const int n = colBase + wn + j * 16 + l16;
            #pragma unroll
            for (int r = 0; r < 8; ++r) {
                const int m = rowBase + wm + i * 16 + r + 8 * g;
                float val = c[i][j][r];
                if (flags & GF_BIAS)  val += bias[n];
                if (flags & GF_RESID) val += resid[(size_t)m * ldr + n];
                if (flags & GF_RELU)  val = fmaxf(val, 0.0f);
                if (flags & GF_OUTT)      Ch[(size_t)n * ldc + m] = (_Float16)val;
                else if (flags & GF_OUTH) Ch[(size_t)m * ldc + n] = (_Float16)val;
                else                      Cf[(size_t)m * ldc + n] = val;
            }
        }
    }
}

// ---------------------------------------------------------------------------
// Embedding lookup + positional encoding
// ---------------------------------------------------------------------------
__global__ __launch_bounds__(TPB)
void embed_kernel(const int* __restrict__ x, const float* __restrict__ emb,
                  const float* __restrict__ pe, float* __restrict__ h,
                  int T, int D, int total)
{
    const int i = blockIdx.x * TPB + threadIdx.x;
    if (i >= total) return;
    const int row = i / D, d = i - row * D;
    const int t = row % T;
    h[i] = emb[(size_t)x[row] * D + d] + pe[(size_t)t * D + d];
}

// ---------------------------------------------------------------------------
// LayerNorm (population variance) -> f16, one block per row
// ---------------------------------------------------------------------------
__global__ __launch_bounds__(TPB)
void layernorm_f16(const float* __restrict__ X, const float* __restrict__ gamma,
                   const float* __restrict__ beta, _Float16* __restrict__ Y, int D)
{
    const int row = blockIdx.x;
    const float* x = X + (size_t)row * D;
    __shared__ float s1[TPB], s2[TPB];
    float a = 0.f, b = 0.f;
    for (int j = threadIdx.x; j < D; j += TPB) { const float v = x[j]; a += v; b += v * v; }
    s1[threadIdx.x] = a; s2[threadIdx.x] = b;
    __syncthreads();
    for (int s = TPB / 2; s > 0; s >>= 1) {
        if (threadIdx.x < s) { s1[threadIdx.x] += s1[threadIdx.x + s]; s2[threadIdx.x] += s2[threadIdx.x + s]; }
        __syncthreads();
    }
    const float mean = s1[0] / D;
    const float var  = s2[0] / D - mean * mean;
    const float rstd = rsqrtf(var + 1e-5f);
    _Float16* y = Y + (size_t)row * D;
    for (int j = threadIdx.x; j < D; j += TPB)
        y[j] = (_Float16)((x[j] - mean) * rstd * gamma[j] + beta[j]);
}

// ---------------------------------------------------------------------------
// Causal softmax over one [T,T] score matrix; f16 probs, masked region zeroed
// ---------------------------------------------------------------------------
__global__ __launch_bounds__(TPB)
void softmax_causal(const float* __restrict__ S, _Float16* __restrict__ P,
                    int T, float scale)
{
    const int t = blockIdx.x;
    const float* row = S + (size_t)t * T;
    _Float16* prow = P + (size_t)t * T;
    __shared__ float red[TPB];
    const int tid = threadIdx.x;
    const int valid = t + 1;

    float mx = -3.0e38f;
    for (int j = tid; j < valid; j += TPB) mx = fmaxf(mx, row[j]);
    red[tid] = mx; __syncthreads();
    for (int s = TPB / 2; s > 0; s >>= 1) {
        if (tid < s) red[tid] = fmaxf(red[tid], red[tid + s]);
        __syncthreads();
    }
    mx = red[0]; __syncthreads();

    float sum = 0.f;
    for (int j = tid; j < valid; j += TPB) sum += __expf(scale * (row[j] - mx));
    red[tid] = sum; __syncthreads();
    for (int s = TPB / 2; s > 0; s >>= 1) {
        if (tid < s) red[tid] += red[tid + s];
        __syncthreads();
    }
    const float inv = 1.0f / red[0];

    for (int j = tid; j < T; j += TPB) {
        const float p = (j < valid) ? __expf(scale * (row[j] - mx)) * inv : 0.0f;
        prow[j] = (_Float16)p;
    }
}

// ---------------------------------------------------------------------------
// f32 -> f16 flat cast
// ---------------------------------------------------------------------------
__global__ __launch_bounds__(TPB)
void cast_f16(const float* __restrict__ in, _Float16* __restrict__ out, int n)
{
    const int i = blockIdx.x * TPB + threadIdx.x;
    if (i < n) out[i] = (_Float16)in[i];
}

// ---------------------------------------------------------------------------
// Batched transpose-cast: in f32 [Z,R,C] -> out f16 [Z,C,R]
// ---------------------------------------------------------------------------
__global__ __launch_bounds__(TPB)
void tcast_f16(const float* __restrict__ in, _Float16* __restrict__ out,
               int R, int C, int total)
{
    const int i = blockIdx.x * TPB + threadIdx.x;
    if (i >= total) return;
    const int z = i / (R * C);
    const int rc = i - z * (R * C);
    const int r = rc / C, c = rc - r * C;
    out[(size_t)z * R * C + (size_t)c * R + r] = (_Float16)in[i];
}

// ---------------------------------------------------------------------------
// Host orchestration
// ---------------------------------------------------------------------------
static inline void launch_gemm(hipStream_t s,
                               const _Float16* A, int lda,
                               const _Float16* B, int ldb,
                               float* Cf, _Float16* Ch, int ldc,
                               const float* bias, const float* resid, int ldr,
                               int M, int N, int K, int flags)
{
    dim3 grid(N / 64, M / 128);
    gemm_wmma<<<grid, TPB, 0, s>>>(A, lda, B, ldb, Cf, Ch, ldc, bias, resid, ldr, K, flags);
}

extern "C" void kernel_launch(void* const* d_in, const int* in_sizes, int n_in,
                              void* d_out, int out_size, void* d_ws, size_t ws_size,
                              hipStream_t stream)
{
    (void)in_sizes; (void)n_in; (void)out_size; (void)ws_size;
    const int V = 32000, D = 1024, H = 16, HD = 64, FF = 4096, L = 4, Bb = 2, T = 1024;
    const int M = Bb * T; // 2048 token rows

    const int*   x   = (const int*)d_in[0];
    const float* emb = (const float*)d_in[1];
    const float* pe  = (const float*)d_in[2];
    const float* lnf_g = (const float*)d_in[3 + 13 * L + 0];
    const float* lnf_b = (const float*)d_in[3 + 13 * L + 1];
    const float* b_u   = (const float*)d_in[3 + 13 * L + 2];

    // ---- workspace carve-up (256B aligned)
    char* ws = (char*)d_ws;
    size_t off = 0;
    auto take = [&](size_t bytes) -> char* {
        char* p = ws + off;
        off = (off + bytes + 255) & ~(size_t)255;
        return p;
    };
    _Float16* embT = (_Float16*)take((size_t)V * D * 2);      // tied unembedding [V, K=D]
    _Float16 *wqT[4], *wkT[4], *wvT[4], *woT[4], *w1T[4], *w2T[4];
    for (int l = 0; l < L; ++l) {
        wqT[l] = (_Float16*)take((size_t)D * D * 2);          // [H*HD, D]
        wkT[l] = (_Float16*)take((size_t)D * D * 2);
        wvT[l] = (_Float16*)take((size_t)D * D * 2);
        woT[l] = (_Float16*)take((size_t)D * D * 2);          // [D, H*HD]
        w1T[l] = (_Float16*)take((size_t)D * FF * 2);         // [FF, D]
        w2T[l] = (_Float16*)take((size_t)FF * D * 2);         // [D, FF]
    }
    float*    hbuf = (float*)take((size_t)M * D * 4);         // residual stream f32
    _Float16* hn   = (_Float16*)take((size_t)M * D * 2);      // LN output f16
    _Float16* qb   = (_Float16*)take((size_t)M * D * 2);      // q  [m, h*HD+hd]
    _Float16* kb   = (_Float16*)take((size_t)M * D * 2);      // k  [m, h*HD+hd]
    _Float16* vt   = (_Float16*)take((size_t)D * M * 2);      // vT [h*HD+hd, m]
    _Float16* ob   = (_Float16*)take((size_t)M * D * 2);      // attn out (concat heads)
    float*    sc   = (float*)take((size_t)T * T * 4);         // scores, reused per (b,h)
    _Float16* pb   = (_Float16*)take((size_t)T * T * 2);      // probs,  reused per (b,h)
    _Float16* ff   = (_Float16*)take((size_t)M * FF * 2);     // FFN hidden f16

    // ---- weight prep (every call; deterministic)
    {
        const int n = V * D;
        cast_f16<<<(n + TPB - 1) / TPB, TPB, 0, stream>>>(emb, embT, n);
    }
    for (int l = 0; l < L; ++l) {
        const int base = 3 + 13 * l;
        const float* Wq = (const float*)d_in[base + 2];
        const float* Wk = (const float*)d_in[base + 3];
        const float* Wv = (const float*)d_in[base + 4];
        const float* Wo = (const float*)d_in[base + 5];
        const float* W1 = (const float*)d_in[base + 9];
        const float* W2 = (const float*)d_in[base + 11];
        const int nqkv = H * D * HD;
        tcast_f16<<<(nqkv + TPB - 1) / TPB, TPB, 0, stream>>>(Wq, wqT[l], D, HD, nqkv);
        tcast_f16<<<(nqkv + TPB - 1) / TPB, TPB, 0, stream>>>(Wk, wkT[l], D, HD, nqkv);
        tcast_f16<<<(nqkv + TPB - 1) / TPB, TPB, 0, stream>>>(Wv, wvT[l], D, HD, nqkv);
        const int ndd = D * D;
        tcast_f16<<<(ndd + TPB - 1) / TPB, TPB, 0, stream>>>(Wo, woT[l], D, D, ndd);
        const int ndf = D * FF;
        tcast_f16<<<(ndf + TPB - 1) / TPB, TPB, 0, stream>>>(W1, w1T[l], D, FF, ndf);
        tcast_f16<<<(ndf + TPB - 1) / TPB, TPB, 0, stream>>>(W2, w2T[l], FF, D, ndf);
    }

    // ---- embedding + PE
    {
        const int total = M * D;
        embed_kernel<<<(total + TPB - 1) / TPB, TPB, 0, stream>>>(x, emb, pe, hbuf, T, D, total);
    }

    // ---- transformer layers
    for (int l = 0; l < L; ++l) {
        const int base = 3 + 13 * l;
        const float* ln1_g = (const float*)d_in[base + 0];
        const float* ln1_b = (const float*)d_in[base + 1];
        const float* bo    = (const float*)d_in[base + 6];
        const float* ln2_g = (const float*)d_in[base + 7];
        const float* ln2_b = (const float*)d_in[base + 8];
        const float* b1    = (const float*)d_in[base + 10];
        const float* b2    = (const float*)d_in[base + 12];

        // pre-LN attention
        layernorm_f16<<<M, TPB, 0, stream>>>(hbuf, ln1_g, ln1_b, hn, D);
        launch_gemm(stream, hn, D, wqT[l], D, nullptr, qb, D, nullptr, nullptr, 0, M, D, D, GF_OUTH);
        launch_gemm(stream, hn, D, wkT[l], D, nullptr, kb, D, nullptr, nullptr, 0, M, D, D, GF_OUTH);
        // v projection written transposed: vt[n, m] so P*V consumes contiguous-K Bt
        launch_gemm(stream, hn, D, wvT[l], D, nullptr, vt, M, nullptr, nullptr, 0, M, D, D, GF_OUTT);

        for (int b = 0; b < Bb; ++b) {
            for (int hh = 0; hh < H; ++hh) {
                const size_t qoff = (size_t)b * T * D + (size_t)hh * HD;
                // scores[T,T] = q[T,HD] * k^T  (k stored [T,HD] == Bt[N=T,K=HD])
                launch_gemm(stream, qb + qoff, D, kb + qoff, D, sc, nullptr, T,
                            nullptr, nullptr, 0, T, T, HD, 0);
                softmax_causal<<<T, TPB, 0, stream>>>(sc, pb, T, 0.125f);
                // o[T,HD] = P[T,T] * v ; Bt = vT[hd, s] slice, ldb = M
                const _Float16* vtb = vt + (size_t)hh * HD * M + (size_t)b * T;
                launch_gemm(stream, pb, T, vtb, M, nullptr, ob + qoff, D,
                            nullptr, nullptr, 0, T, HD, T, GF_OUTH);
            }
        }
        // h = h + o @ Wo + bo   (in-place residual)
        launch_gemm(stream, ob, D, woT[l], D, hbuf, nullptr, D, bo, hbuf, D,
                    M, D, D, GF_BIAS | GF_RESID);

        // pre-LN FFN
        layernorm_f16<<<M, TPB, 0, stream>>>(hbuf, ln2_g, ln2_b, hn, D);
        launch_gemm(stream, hn, D, w1T[l], D, nullptr, ff, FF, b1, nullptr, 0,
                    M, FF, D, GF_BIAS | GF_RELU | GF_OUTH);
        launch_gemm(stream, ff, FF, w2T[l], FF, hbuf, nullptr, D, b2, hbuf, D,
                    M, D, FF, GF_BIAS | GF_RESID);
    }

    // ---- final LN + tied unembedding
    layernorm_f16<<<M, TPB, 0, stream>>>(hbuf, lnf_g, lnf_b, hn, D);
    launch_gemm(stream, hn, D, embT, D, (float*)d_out, nullptr, V, b_u, nullptr, 0,
                M, V, D, GF_BIAS);
}
